// OTRobertaSelfAttention_78365973283445
// MI455X (gfx1250) — matrix operand order repaired and checked by reading, not compiled, not run on hardware
//
#include <hip/hip_runtime.h>
#include <hip/hip_bf16.h>
#include <math.h>

// ---------------- sizes ----------------
#define BB   8
#define SS   512
#define EE   1024
#define HH   16
#define DD   64
#define BH   (BB*HH)          // 128
#define NROW (BB*SS)          // 4096
#define NELT ((size_t)BH*SS*DD) // 4,194,304 elements per q/k/v tensor

// ---------------- WMMA fragment types ----------------
typedef _Float16 v16h __attribute__((ext_vector_type(16)));
typedef float    v8f  __attribute__((ext_vector_type(8)));
typedef unsigned v4u  __attribute__((ext_vector_type(4)));

union FragU { v4u u[2]; v16h h; };

__device__ __forceinline__ v16h load_pair(const _Float16* p0, const _Float16* p1) {
  FragU f;
  f.u[0] = *(const v4u*)p0;
  f.u[1] = *(const v4u*)p1;
  return f.h;
}

// A fragment: 16x32 f16, row-major source. lane holds row M=lane&15,
// K = [g*8, g*8+8) and [16+g*8, 16+g*8+8), g = lane>>4.
__device__ __forceinline__ v16h load_a_frag(const _Float16* base, int lda) {
  int lane = threadIdx.x & 31;
  const _Float16* row = base + (lane & 15) * lda + ((lane >> 4) << 3);
  return load_pair(row, row + 16);
}

// B fragment: 32x16 f16, K-major source (B stored as [K][N] row-major).
// lane holds N=lane&15, K = [g*8, g*8+8) and [16+g*8, 16+g*8+8), g = lane>>4.
__device__ __forceinline__ v16h load_b_frag(const _Float16* base, int ldb) {
  int lane = threadIdx.x & 31;
  const _Float16* col = base + (lane & 15) + ((lane >> 4) << 3) * ldb;
  v16h b;
  #pragma unroll
  for (int s = 0; s < 8; s++) {
    b[s]     = col[s * ldb];
    b[8 + s] = col[(16 + s) * ldb];
  }
  return b;
}

__device__ __forceinline__ v8f wmma_f16(v16h a, v16h b, v8f c) {
  return __builtin_amdgcn_wmma_f32_16x16x32_f16(false, a, false, b, (short)0, c, false, false);
}

// ---------------- ordered-uint float min/max atomics ----------------
__device__ __forceinline__ unsigned fmap(float f) {
  unsigned u = __float_as_uint(f);
  return (u >> 31) ? ~u : (u | 0x80000000u);
}
__device__ __forceinline__ float funmap(unsigned m) {
  return (m & 0x80000000u) ? __uint_as_float(m & 0x7FFFFFFFu) : __uint_as_float(~m);
}
// slots: 0,1=q  2,3=k  4,5=v  6,7=qk  8,9=attn  10,11=av  (even=min, odd=max)
__device__ __forceinline__ void wave_minmax_atomic(unsigned* stats, int slot, float mn, float mx) {
  #pragma unroll
  for (int off = 16; off; off >>= 1) {
    mn = fminf(mn, __shfl_xor(mn, off, 32));
    mx = fmaxf(mx, __shfl_xor(mx, off, 32));
  }
  if ((threadIdx.x & 31) == 0) {
    atomicMin(&stats[slot],     fmap(mn));
    atomicMax(&stats[slot + 1], fmap(mx));
  }
}

// ---------------- quantizers (forward value of straight-through) ----------------
__device__ __forceinline__ float uni_q(float x, float mn, float mx) {
  float sc = (mx - mn) * (1.0f / 255.0f);
  float xc = fminf(fmaxf(x, mn), mx);
  return rintf((xc - mn) / sc) * sc + mn;   // rint == round-half-even, matches jnp.round
}
__device__ __forceinline__ float lut_q(float x, float absmax) {
  const float lmin   = 0.02004f;
  const float logmin = __logf(lmin);
  const float step   = -logmin * (1.0f / 255.0f);
  float mag  = fminf(fmaxf(fabsf(x) / absmax, lmin), 1.0f);
  float idx  = rintf((__logf(mag) - logmin) / step);
  float qmag = __expf(logmin + idx * step) * absmax;
  return (fabsf(x) < 0.5f * lmin * absmax) ? 0.0f : copysignf(qmag, x);
}

// ================= K0: stats init + f32->f16 + weight transpose =================
__global__ __launch_bounds__(256) void prep_kernel(
    const float* __restrict__ hs, const float* __restrict__ Wq,
    const float* __restrict__ Wk, const float* __restrict__ Wv,
    _Float16* __restrict__ hs16, _Float16* __restrict__ wtq,
    _Float16* __restrict__ wtk,  _Float16* __restrict__ wtv,
    unsigned* __restrict__ stats) {
  size_t gid = (size_t)blockIdx.x * blockDim.x + threadIdx.x;
  if (gid < 16) stats[gid] = (gid & 1) ? 0u : 0xFFFFFFFFu;
  const size_t NH = (size_t)NROW * EE;   // 4M
  const size_t NW = (size_t)EE * EE;     // 1M per weight
  size_t total  = NH + 3 * NW;
  size_t stride = (size_t)gridDim.x * blockDim.x;
  for (size_t i = gid; i < total; i += stride) {
    if (i < NH) {
      hs16[i] = (_Float16)hs[i];
    } else {
      size_t j = i - NH;
      int wsel = (int)(j / NW);
      size_t r = j % NW;                  // r = k*1024 + n  (K-major dest)
      int k = (int)(r >> 10), n = (int)(r & 1023);
      const float* W  = (wsel == 0) ? Wq : (wsel == 1) ? Wk : Wv;
      _Float16*    wt = (wsel == 0) ? wtq : (wsel == 1) ? wtk : wtv;
      wt[r] = (_Float16)W[(size_t)n * EE + k];  // wt[k][n] = W[n][k]
    }
  }
}

// ================= K1: QKV projection (NT GEMM, f16 WMMA) + q/k/v minmax =======
// grid (64 n-tiles, 32, z=3), block 256 (8 waves; wave -> m-tile)
__global__ __launch_bounds__(256) void proj_kernel(
    const _Float16* __restrict__ hs16,
    const _Float16* __restrict__ wtq, const _Float16* __restrict__ wtk,
    const _Float16* __restrict__ wtv,
    const float* __restrict__ bq, const float* __restrict__ bk,
    const float* __restrict__ bv,
    _Float16* __restrict__ qh, _Float16* __restrict__ kt,
    _Float16* __restrict__ vh, unsigned* __restrict__ stats) {
  int z = blockIdx.z;
  const _Float16* wt   = (z == 0) ? wtq : (z == 1) ? wtk : wtv;
  const float*    bias = (z == 0) ? bq  : (z == 1) ? bk  : bv;
  int wave = threadIdx.x >> 5, lane = threadIdx.x & 31;
  int g = lane >> 4, ln = lane & 15;
  int mt = blockIdx.y * 8 + wave;   // 0..255
  int nt = blockIdx.x;              // 0..63
  const _Float16* A  = hs16 + (size_t)(mt * 16) * EE;
  const _Float16* Bm = wt + nt * 16;            // K-major, ld = EE
  v8f c = {0.f,0.f,0.f,0.f,0.f,0.f,0.f,0.f};
  for (int k0 = 0; k0 < EE; k0 += 32) {
    v16h a  = load_a_frag(A + k0, EE);
    v16h bf = load_b_frag(Bm + (size_t)k0 * EE, EE);
    c = wmma_f16(a, bf, c);
  }
  float bval = bias[nt * 16 + ln];
  float mn = 1e38f, mx = -1e38f;
  #pragma unroll
  for (int r = 0; r < 8; r++) {
    c[r] += bval;
    mn = fminf(mn, c[r]); mx = fmaxf(mx, c[r]);
  }
  wave_minmax_atomic(stats, 2 * z, mn, mx);
  // store with head split; K stored transposed ([BH,D,S]) for later K-major use
  int gn = nt * 16 + ln;
  int h = gn >> 6, d = gn & 63;
  #pragma unroll
  for (int r = 0; r < 8; r++) {
    int gm = mt * 16 + r + 8 * g;
    int b = gm >> 9, s = gm & 511;
    int bh = b * HH + h;
    _Float16 val = (_Float16)c[r];
    if (z == 1) kt[((size_t)bh * DD + d) * SS + s] = val;
    else {
      _Float16* o = (z == 0) ? qh : vh;
      o[((size_t)bh * SS + s) * DD + d] = val;
    }
  }
}

// ================= K2: full-precision QK^T min/max only (nothing stored) ========
// grid (32 row-tiles, 128 bh), block 256: wave w handles col tiles 4w..4w+3
__global__ __launch_bounds__(256) void qkstat_kernel(
    const _Float16* __restrict__ qh, const _Float16* __restrict__ kt,
    unsigned* __restrict__ stats) {
  int bh = blockIdx.y;
  int s0 = blockIdx.x * 16;
  int wave = threadIdx.x >> 5;
  const _Float16* A = qh + ((size_t)bh * SS + s0) * DD;
  const _Float16* B = kt + (size_t)bh * DD * SS;   // K-major [D][S]
  v8f zero = {0.f,0.f,0.f,0.f,0.f,0.f,0.f,0.f};
  v8f c[4] = {zero, zero, zero, zero};
  for (int k0 = 0; k0 < DD; k0 += 32) {
    v16h a = load_a_frag(A + k0, DD);
    #pragma unroll
    for (int t = 0; t < 4; t++) {
      int n0 = (wave * 4 + t) * 16;
      v16h bf = load_b_frag(B + (size_t)k0 * SS + n0, SS);
      c[t] = wmma_f16(a, bf, c[t]);
    }
  }
  float mn = 1e38f, mx = -1e38f;
  #pragma unroll
  for (int t = 0; t < 4; t++)
    #pragma unroll
    for (int r = 0; r < 8; r++) { mn = fminf(mn, c[t][r]); mx = fmaxf(mx, c[t][r]); }
  wave_minmax_atomic(stats, 6, mn, mx);
}

// ================= K3: elementwise quantization of q (uniform), k,v (LUT) =======
__global__ __launch_bounds__(256) void quant_kernel(
    const _Float16* __restrict__ qh, const _Float16* __restrict__ kt,
    const _Float16* __restrict__ vh,
    _Float16* __restrict__ aq, _Float16* __restrict__ kq,
    _Float16* __restrict__ vq, const unsigned* __restrict__ stats) {
  float qmn = funmap(stats[0]), qmx = funmap(stats[1]);
  float kabs = fmaxf(fabsf(funmap(stats[2])), fabsf(funmap(stats[3])));
  float vabs = fmaxf(fabsf(funmap(stats[4])), fabsf(funmap(stats[5])));
  size_t gid = (size_t)blockIdx.x * blockDim.x + threadIdx.x;
  size_t stride = (size_t)gridDim.x * blockDim.x;
  for (size_t i = gid; i < 3 * NELT; i += stride) {
    if (i < NELT)            aq[i]            = (_Float16)uni_q((float)qh[i], qmn, qmx);
    else if (i < 2 * NELT)   kq[i - NELT]     = (_Float16)lut_q((float)kt[i - NELT], kabs);
    else                     vq[i - 2 * NELT] = (_Float16)lut_q((float)vh[i - 2 * NELT], vabs);
  }
}

// ================= K4/K5: fused quantized-QK^T + softmax + AV ===================
// mode 0: raw probs + unquantized v -> attn/av min-max stats (atomics)
// mode 1: uniform_q(probs) @ lut_q(v) -> uniform_q -> context output
// grid (32 row-tiles, 128 bh), block 256 (8 waves)
__global__ __launch_bounds__(256) void attn_kernel(
    const _Float16* __restrict__ aq,   // [BH,S,D] quantized q
    const _Float16* __restrict__ kqt,  // [BH,D,S] quantized k^T (K-major)
    const _Float16* __restrict__ vmat, // [BH,S,D] (raw v for mode0, quantized for mode1)
    const float* __restrict__ mask,    // [B,S] additive
    unsigned* __restrict__ stats,
    float* __restrict__ out,           // [B,S,E]
    int mode) {
  __shared__ __align__(16) float    sc[16][SS];
  __shared__ __align__(16) _Float16 pr[16][SS];
  __shared__ float red[16][16];
  __shared__ float rowv[16];

  int bh = blockIdx.y;
  int b = bh >> 4, h = bh & 15;
  int s0 = blockIdx.x * 16;
  int tid = threadIdx.x;
  int wave = tid >> 5, lane = tid & 31;
  int g = lane >> 4, ln = lane & 15;

  float qkmn = funmap(stats[6]), qkmx = funmap(stats[7]);

  // ---- 1. quantized scores for this 16x512 stripe ----
  const _Float16* A = aq + ((size_t)bh * SS + s0) * DD;
  const _Float16* B = kqt + (size_t)bh * DD * SS;
  v8f zero = {0.f,0.f,0.f,0.f,0.f,0.f,0.f,0.f};
  v8f c[4] = {zero, zero, zero, zero};
  for (int k0 = 0; k0 < DD; k0 += 32) {
    v16h a = load_a_frag(A + k0, DD);
    #pragma unroll
    for (int t = 0; t < 4; t++) {
      int n0 = (wave * 4 + t) * 16;
      v16h bf = load_b_frag(B + (size_t)k0 * SS + n0, SS);
      c[t] = wmma_f16(a, bf, c[t]);
    }
  }
  #pragma unroll
  for (int t = 0; t < 4; t++) {
    int col = (wave * 4 + t) * 16 + ln;
    float mk = mask[(size_t)b * SS + col];
    #pragma unroll
    for (int r = 0; r < 8; r++)
      sc[r + 8 * g][col] = uni_q(c[t][r], qkmn, qkmx) * 0.125f + mk;
  }
  __syncthreads();

  // ---- 2. row max ----
  {
    int row = tid >> 4, ch = tid & 15;
    float m = -1e38f;
    for (int j = 0; j < 32; j++) m = fmaxf(m, sc[row][ch * 32 + j]);
    red[row][ch] = m;
  }
  __syncthreads();
  if (tid < 16) {
    float m = red[tid][0];
    for (int j = 1; j < 16; j++) m = fmaxf(m, red[tid][j]);
    rowv[tid] = m;
  }
  __syncthreads();

  // ---- 3. exp + row sum ----
  {
    int row = tid >> 4, ch = tid & 15;
    float rm = rowv[row], sum = 0.f;
    for (int j = 0; j < 32; j++) {
      float e = __expf(sc[row][ch * 32 + j] - rm);
      sc[row][ch * 32 + j] = e;
      sum += e;
    }
    red[row][ch] = sum;
  }
  __syncthreads();
  if (tid < 16) {
    float s = 0.f;
    for (int j = 0; j < 16; j++) s += red[tid][j];
    rowv[tid] = s;
  }
  __syncthreads();

  // ---- 4. normalize -> probs (quantize in mode 1; minmax stats in mode 0) ----
  float amn = 0.f, amx = 1.f;
  if (mode == 1) {  // attn stats start finite [0,1] -> smoothed update
    amn = 0.1f * funmap(stats[8]);
    amx = 0.9f + 0.1f * funmap(stats[9]);
  }
  float pmn = 1e38f, pmx = -1e38f;
  {
    int row = tid >> 4, ch = tid & 15;
    float inv = 1.f / rowv[row];
    for (int j = 0; j < 32; j++) {
      float p = sc[row][ch * 32 + j] * inv;
      if (mode == 0) { pmn = fminf(pmn, p); pmx = fmaxf(pmx, p); }
      else           { p = uni_q(p, amn, amx); }
      pr[row][ch * 32 + j] = (_Float16)p;
    }
  }
  if (mode == 0) wave_minmax_atomic(stats, 8, pmn, pmx);
  __syncthreads();

  // ---- 5. AV: waves 0..3 each produce one 16x16 tile of the 16x64 context ----
  if (wave < 4) {
    v8f cc = zero;
    const _Float16* Vb = vmat + (size_t)bh * SS * DD + wave * 16; // K-major [S][D]
    for (int k0 = 0; k0 < SS; k0 += 32) {
      v16h a  = load_a_frag(&pr[0][0] + k0, SS);
      v16h bf = load_b_frag(Vb + (size_t)k0 * DD, DD);
      cc = wmma_f16(a, bf, cc);
    }
    if (mode == 0) {
      float mn = cc[0], mx = cc[0];
      #pragma unroll
      for (int r = 1; r < 8; r++) { mn = fminf(mn, cc[r]); mx = fmaxf(mx, cc[r]); }
      wave_minmax_atomic(stats, 10, mn, mx);
    } else {
      float avmn = funmap(stats[10]), avmx = funmap(stats[11]);
      #pragma unroll
      for (int r = 0; r < 8; r++) {
        int m = r + 8 * g;
        int d = wave * 16 + ln;
        out[((size_t)(b * SS + s0 + m)) * EE + h * DD + d] = uni_q(cc[r], avmn, avmx);
      }
    }
  }
}

// ================================ launch ========================================
extern "C" void kernel_launch(void* const* d_in, const int* in_sizes, int n_in,
                              void* d_out, int out_size, void* d_ws, size_t ws_size,
                              hipStream_t stream) {
  (void)in_sizes; (void)n_in; (void)out_size; (void)ws_size;
  const float* hs   = (const float*)d_in[0];
  const float* mask = (const float*)d_in[1];
  const float* Wq   = (const float*)d_in[2];
  const float* bq   = (const float*)d_in[3];
  const float* Wk   = (const float*)d_in[4];
  const float* bk   = (const float*)d_in[5];
  const float* Wv   = (const float*)d_in[6];
  const float* bv   = (const float*)d_in[7];
  float* out = (float*)d_out;

  // workspace carve-up (~62.3 MB total)
  char* w = (char*)d_ws;
  unsigned* stats = (unsigned*)w;          w += 256;
  _Float16* hs16 = (_Float16*)w;           w += (size_t)NROW * EE * 2;  // 8 MB
  _Float16* wtq  = (_Float16*)w;           w += (size_t)EE * EE * 2;    // 2 MB
  _Float16* wtk  = (_Float16*)w;           w += (size_t)EE * EE * 2;
  _Float16* wtv  = (_Float16*)w;           w += (size_t)EE * EE * 2;
  _Float16* qh   = (_Float16*)w;           w += NELT * 2;               // 8 MB each
  _Float16* kt   = (_Float16*)w;           w += NELT * 2;
  _Float16* vh   = (_Float16*)w;           w += NELT * 2;
  _Float16* aq   = (_Float16*)w;           w += NELT * 2;
  _Float16* kq   = (_Float16*)w;           w += NELT * 2;
  _Float16* vq   = (_Float16*)w;           w += NELT * 2;

  // K0: stats init + precision convert + weight transpose (K-major f16)
  prep_kernel<<<4096, 256, 0, stream>>>(hs, Wq, Wk, Wv, hs16, wtq, wtk, wtv, stats);
  // K1: QKV projections (WMMA) + q/k/v min-max
  proj_kernel<<<dim3(64, 32, 3), 256, 0, stream>>>(hs16, wtq, wtk, wtv,
                                                   bq, bk, bv, qh, kt, vh, stats);
  // K2: full-precision score min-max (never materialized)
  qkstat_kernel<<<dim3(32, BH), 256, 0, stream>>>(qh, kt, stats);
  // K3: quantize q (uniform) / k,v (LUT)
  quant_kernel<<<4096, 256, 0, stream>>>(qh, kt, vh, aq, kq, vq, stats);
  // K4: fused quantized-QK + softmax -> attn & av stats (raw v)
  attn_kernel<<<dim3(32, BH), 256, 0, stream>>>(aq, kq, vh, mask, stats, out, 0);
  // K5: fused recompute + quantized AV -> context
  attn_kernel<<<dim3(32, BH), 256, 0, stream>>>(aq, kq, vq, mask, stats, out, 1);
}
